// SpatialSelectiveScan2D_6579889897506
// MI455X (gfx1250) — compile-verified, matrix operand order checked by reading, not run on hardware
//
#include <hip/hip_runtime.h>
#include <hip/hip_bf16.h>

// ---------------------------------------------------------------------------
// SpatialSelectiveScan2D for MI455X (gfx1250, wave32, WMMA)
//
// ~36 GFLOP / ~1.5 GB traffic -> bandwidth bound at 23.3 TB/s. All dense
// per-token matmuls on v_wmma_f32_16x16x32_f16 (f32 acc, f16 operands).
// Weights pre-converted once to f16 (row length % 8 == 0) so an A-fragment
// is two 16B global loads; activations staged in LDS column-major
// ([token][K], ld % 8 == 0) so a B-fragment is two ds_load_b128.
// Activations use the TRANS pipe (v_exp_f32 + v_rcp_f32). The serial
// recurrence runs as 512 independent wave32 scans with shuffle-based
// affine-map composition. Direction permutations folded into gathers.
// ---------------------------------------------------------------------------

#define DEVFN __device__ __forceinline__

constexpr int B_   = 4;
constexpr int DIM_ = 128;
constexpr int H_   = 96;
constexpr int W_   = 96;
constexpr int L_   = H_ * W_;      // 9216
constexpr int HID_ = 256;
constexpr int MH_  = 32;
constexpr int KCONV = 5;
constexpr int NSEQ = 16;           // 4 directions * B
constexpr int TN   = 48;           // token tile (192 tiles over L)
constexpr float DMIN_ = 0.999f;
constexpr float DMAX_ = 0.9999f;

// padded LDS leading dims (multiples of 8 -> 16B-aligned v8h runs)
constexpr int LD_DIM = DIM_ + 8;        // 136
constexpr int LD_HID = HID_ + 8;        // 264
constexpr int LD_MH  = MH_ + 8;         // 40
constexpr int LD_CAT = 4 * DIM_ + 8;    // 520

typedef __attribute__((ext_vector_type(16))) _Float16 v16h;
typedef __attribute__((ext_vector_type(8)))  _Float16 v8h;
typedef __attribute__((ext_vector_type(8)))  float    v8f;

DEVFN int lane_id() { return (int)(threadIdx.x & 31u); }

// Fragment K-packing (ISA 7.12.2): a lane's 16 elements are two contiguous
// 8-element K-runs: k = half*8 + e (e<8) and k = 16 + half*8 + (e-8).

// A fragment (16x32, MxK) from row-major f16 weights (ldw % 8 == 0).
DEVFN v16h frag_a_f16g(const _Float16* __restrict__ W, int ldw, int r0, int c0) {
  int l = lane_id(); int m = l & 15, half = l >> 4;
  const _Float16* base = W + (size_t)(r0 + m) * ldw + c0 + half * 8;
  v8h lo = *(const v8h*)(base);
  v8h hi = *(const v8h*)(base + 16);
  v16h a;
#pragma unroll
  for (int e = 0; e < 8; ++e) { a[e] = lo[e]; a[e + 8] = hi[e]; }
  return a;
}

// B fragment (32x16, KxN) from column-major f16 LDS tile Bm[token][k]
// (ld = ldk, multiple of 8). Two ds_load_b128 per lane.
DEVFN v16h frag_b_ldsT(const _Float16* Bm, int ldk, int k0, int n0) {
  int l = lane_id(); int n = l & 15, half = l >> 4;
  const _Float16* base = Bm + (size_t)(n0 + n) * ldk + k0 + half * 8;
  v8h lo = *(const v8h*)(base);
  v8h hi = *(const v8h*)(base + 16);
  v16h b;
#pragma unroll
  for (int e = 0; e < 8; ++e) { b[e] = lo[e]; b[e + 8] = hi[e]; }
  return b;
}

DEVFN v8f wmma_f16(v16h a, v16h b, v8f c) {
  return __builtin_amdgcn_wmma_f32_16x16x32_f16(false, a, false, b,
                                                (short)0, c, false, false);
}

// 8 consecutive f32 (16B-aligned base) -> two b128 loads.
DEVFN void load_bias8(const float* __restrict__ p, float out[8]) {
  float4 a = *(const float4*)p;
  float4 b = *(const float4*)(p + 4);
  out[0] = a.x; out[1] = a.y; out[2] = a.z; out[3] = a.w;
  out[4] = b.x; out[5] = b.y; out[6] = b.z; out[7] = b.w;
}

// TRANS-pipe activations: v_exp_f32 + v_rcp_f32, no IEEE division.
DEVFN float frcp(float x) { return __builtin_amdgcn_rcpf(x); }
DEVFN float sigm(float v) { return frcp(1.0f + __expf(-v)); }
DEVFN float fast_tanh(float v) { return 1.0f - 2.0f * frcp(1.0f + __expf(2.0f * v)); }

// token l of direction-sequence `dir` -> flat row-major pixel index
DEVFN int dir_src_index(int dir, int l) {
  int q = (dir & 1) ? (L_ - 1 - l) : l;
  if (dir >= 2) { int r = q / H_; int c = q % H_; return c * W_ + r; }
  return q;
}

// ---------------------------------------------------------------------------
// Weight f32 -> f16 conversion (once per launch; weights then L2-resident
// at half the bytes).
// ---------------------------------------------------------------------------
__global__ void kCvt(const float* __restrict__ src, _Float16* __restrict__ dst,
                     int n) {
  int i = (int)(blockIdx.x * blockDim.x + threadIdx.x);
  if (i < n) dst[i] = (_Float16)src[i];
}

// ---------------------------------------------------------------------------
// Kernel A1: gather x, grouped in_proj (128x32 per group, K=32 = 1 WMMA),
// SiLU -> u_pre (f16, token-major), sigmoid -> gate (f16, token-major).
// ---------------------------------------------------------------------------
__global__ void kA1_inproj(const float* __restrict__ x,
                           const _Float16* __restrict__ Wp,   // (512,32) f16
                           const float* __restrict__ bp,      // (512)
                           _Float16* __restrict__ upre,       // [NSEQ][L][HID]
                           _Float16* __restrict__ gate) {     // [NSEQ][L][HID]
  const int tile = blockIdx.x, seq = blockIdx.y;
  const int dir = seq / B_, bi = seq % B_;
  const int t0 = tile * TN;
  __shared__ __align__(16) _Float16 xs[TN][LD_DIM];   // [token][channel]

  if (threadIdx.x == 0) { __builtin_prefetch(Wp, 0, 2); __builtin_prefetch(bp, 0, 2); }

  for (int idx = threadIdx.x; idx < TN * DIM_; idx += blockDim.x) {
    int col = idx / DIM_, ch = idx % DIM_;
    int p = dir_src_index(dir, t0 + col);
    xs[col][ch] = (_Float16)x[((size_t)bi * DIM_ + ch) * L_ + p];
  }
  __syncthreads();

  const int wave = threadIdx.x >> 5;
  const int lane = lane_id(), nl = lane & 15, half = lane >> 4;
  for (int t = wave; t < 32 * 3; t += 8) {       // 32 row-tiles x 3 col-tiles
    int rt = t / 3, ct = t % 3;
    int g  = rt >> 3;                            // 128 rows (8 tiles) / group
    v16h a = frag_a_f16g(Wp, 32, rt * 16, 0);
    v16h b = frag_b_ldsT(&xs[0][0], LD_DIM, 32 * g, ct * 16);
    v8f  c = {};
    c = wmma_f16(a, b, c);

    float bias[8];
    load_bias8(bp + rt * 16 + half * 8, bias);
    size_t tokb = (size_t)seq * L_ + t0 + ct * 16 + nl;
    if (rt < 16) {                               // u rows: SiLU (uniform branch)
      int row0 = rt * 16 + half * 8;
#pragma unroll
      for (int i = 0; i < 8; ++i) {
        float v = c[i] + bias[i];
        upre[tokb * HID_ + row0 + i] = (_Float16)(v * sigm(v));
      }
    } else {                                     // gate rows: sigmoid
      int row0 = rt * 16 + half * 8 - HID_;
#pragma unroll
      for (int i = 0; i < 8; ++i) {
        float v = c[i] + bias[i];
        gate[tokb * HID_ + row0 + i] = (_Float16)sigm(v);
      }
    }
  }
}

// ---------------------------------------------------------------------------
// Kernel A2: depthwise conv K=5 (zero-padded halo), then the four 32x256 SSM
// projections stacked into one 128x256 GEMM (K=256 -> 8 WMMA k-steps).
// ---------------------------------------------------------------------------
__global__ void kA2_conv_ssm(const _Float16* __restrict__ upre, // [NSEQ][L][HID]
                             const float* __restrict__ dww,     // (HID,1,5)
                             const _Float16* __restrict__ Wi,   // (32,256) f16
                             const _Float16* __restrict__ Wd, const float* __restrict__ db,
                             const _Float16* __restrict__ Wb, const float* __restrict__ bbb,
                             const _Float16* __restrict__ Wc, const float* __restrict__ cb,
                             _Float16* __restrict__ uc,         // [NSEQ][L][HID]
                             float* __restrict__ sS, float* __restrict__ aS,
                             float* __restrict__ bbS, float* __restrict__ ccS) {
  const int tile = blockIdx.x, seq = blockIdx.y;
  const int t0 = tile * TN;
  __shared__ __align__(16) _Float16 us[TN + 4][LD_HID];   // halo +-2, [token][ch]
  __shared__ __align__(16) _Float16 ucs[TN][LD_HID];

  if (threadIdx.x == 0) {
    __builtin_prefetch(Wi, 0, 2); __builtin_prefetch(Wd, 0, 2);
    __builtin_prefetch(Wb, 0, 2); __builtin_prefetch(Wc, 0, 2);
    __builtin_prefetch(dww, 0, 2);
  }

  for (int idx = threadIdx.x; idx < (TN + 4) * HID_; idx += blockDim.x) {
    int col = idx / HID_, r = idx % HID_;
    int l = t0 + col - 2;
    us[col][r] = (l >= 0 && l < L_)
                     ? upre[((size_t)seq * L_ + l) * HID_ + r]
                     : (_Float16)0.0f;
  }
  __syncthreads();

  for (int idx = threadIdx.x; idx < TN * HID_; idx += blockDim.x) {
    int col = idx / HID_, r = idx % HID_;
    float acc = 0.f;
#pragma unroll
    for (int k = 0; k < KCONV; ++k)
      acc = fmaf(dww[r * KCONV + k], (float)us[col + k][r], acc);
    ucs[col][r] = (_Float16)acc;
    uc[((size_t)seq * L_ + t0 + col) * HID_ + r] = (_Float16)acc;
  }
  __syncthreads();

  const int wave = threadIdx.x >> 5;
  const int lane = lane_id(), nl = lane & 15, half = lane >> 4;
  for (int t = wave; t < 8 * 3; t += 8) {      // 8 row-tiles (4 mats) x 3
    int rt = t / 3, ct = t % 3;
    int wsel = rt >> 1, lr0 = (rt & 1) * 16;
    const _Float16* W = (wsel == 0) ? Wi : (wsel == 1) ? Wd
                      : (wsel == 2) ? Wb : Wc;
    v8f c = {};
#pragma unroll
    for (int ks = 0; ks < HID_ / 32; ++ks) {
      v16h a = frag_a_f16g(W, HID_, lr0, ks * 32);
      v16h b = frag_b_ldsT(&ucs[0][0], LD_HID, ks * 32, ct * 16);
      c = wmma_f16(a, b, c);
    }

    int ch0 = lr0 + half * 8;                  // 8 consecutive channels
    size_t ob = ((size_t)seq * MH_ + ch0) * L_ + t0 + ct * 16 + nl;
    if (wsel == 0) {                           // s: raw
#pragma unroll
      for (int i = 0; i < 8; ++i) sS[ob + (size_t)i * L_] = c[i];
    } else if (wsel == 1) {                    // a
      float bias[8]; load_bias8(db + ch0, bias);
#pragma unroll
      for (int i = 0; i < 8; ++i) {
        float av = DMIN_ + (DMAX_ - DMIN_) * sigm(c[i] + bias[i]);
        aS[ob + (size_t)i * L_] = fminf(fmaxf(av, 1e-4f), 0.9999f);
      }
    } else if (wsel == 2) {                    // bb
      float bias[8]; load_bias8(bbb + ch0, bias);
#pragma unroll
      for (int i = 0; i < 8; ++i)
        bbS[ob + (size_t)i * L_] = fast_tanh(c[i] + bias[i]);
    } else {                                   // cc
      float bias[8]; load_bias8(cb + ch0, bias);
#pragma unroll
      for (int i = 0; i < 8; ++i)
        ccS[ob + (size_t)i * L_] = fast_tanh(c[i] + bias[i]);
    }
  }
}

// ---------------------------------------------------------------------------
// Kernel B: 512 independent scans h_t = a_t*h_{t-1} + bb_t*s_t (one wave32
// each). Per 32-token chunk: inclusive scan of affine maps via __shfl_up
// composition, carry via lane-31 broadcast. Emits g = cc*h + d*s.
// ---------------------------------------------------------------------------
__global__ void kB_scan(const float* __restrict__ sS, const float* __restrict__ aS,
                        const float* __restrict__ bbS, const float* __restrict__ ccS,
                        const float* __restrict__ dvec, float* __restrict__ gS) {
  int gwave = (int)((blockIdx.x * blockDim.x + threadIdx.x) >> 5);
  int lane = lane_id();
  if (gwave >= NSEQ * MH_) return;
  int ch = gwave % MH_;
  size_t base = (size_t)gwave * L_;            // layout [seq][ch][L]
  float dv = dvec[ch];
  float h = 0.0f;
  for (int c0 = 0; c0 < L_; c0 += 32) {
    size_t idx = base + c0 + lane;
    float sv = sS[idx];
    float A  = aS[idx];
    float Bc = bbS[idx] * sv;
#pragma unroll
    for (int off = 1; off < 32; off <<= 1) {
      float Ap = __shfl_up(A, off);
      float Bp = __shfl_up(Bc, off);
      if (lane >= off) { Bc = fmaf(A, Bp, Bc); A = A * Ap; }
    }
    float hl = fmaf(A, h, Bc);
    gS[idx] = fmaf(ccS[idx], hl, dv * sv);
    h = __shfl(hl, 31);
  }
}

// ---------------------------------------------------------------------------
// Kernel C: u2 = uc + ssm_out_w @ g ; v = u2 * gate ; grouped mix (+bias) ;
// out_proj (+bias) -> y_all (token-major). Mix output reuses ucs LDS.
// ---------------------------------------------------------------------------
__global__ void kC_backend(const _Float16* __restrict__ uc,    // [NSEQ][L][HID]
                           const _Float16* __restrict__ gate,  // [NSEQ][L][HID]
                           const float* __restrict__ gS,       // [NSEQ][MH][L]
                           const _Float16* __restrict__ Wout,  // (256,32) f16
                           const _Float16* __restrict__ Wmix,  // (256,64) f16
                           const float* __restrict__ bmix,
                           const _Float16* __restrict__ Wop,   // (128,256) f16
                           const float* __restrict__ bop,
                           float* __restrict__ yAll) {         // [NSEQ][L][DIM]
  const int tile = blockIdx.x, seq = blockIdx.y;
  const int t0 = tile * TN;
  __shared__ __align__(16) _Float16 ucs[TN][LD_HID];  // reused for mix output
  __shared__ __align__(16) _Float16 gs[TN][LD_MH];
  __shared__ __align__(16) _Float16 vs[TN][LD_HID];

  if (threadIdx.x == 0) {
    __builtin_prefetch(Wout, 0, 2); __builtin_prefetch(Wmix, 0, 2);
    __builtin_prefetch(Wop, 0, 2);
  }

  for (int idx = threadIdx.x; idx < TN * HID_; idx += blockDim.x) {
    int col = idx / HID_, r = idx % HID_;
    ucs[col][r] = uc[((size_t)seq * L_ + t0 + col) * HID_ + r];
  }
  for (int idx = threadIdx.x; idx < TN * MH_; idx += blockDim.x) {
    int col = idx / MH_, ch = idx % MH_;
    gs[col][ch] = (_Float16)gS[((size_t)seq * MH_ + ch) * L_ + t0 + col];
  }
  __syncthreads();

  const int wave = threadIdx.x >> 5;
  const int lane = lane_id(), nl = lane & 15, half = lane >> 4;

  // stage 1: u2 = Wout @ g + uc ; v = u2 * gate   (16 rt x 3 ct, K=32)
  for (int t = wave; t < 16 * 3; t += 8) {
    int rt = t / 3, ct = t % 3;
    v16h a = frag_a_f16g(Wout, MH_, rt * 16, 0);
    v16h b = frag_b_ldsT(&gs[0][0], LD_MH, 0, ct * 16);
    v8f  c = {};
    c = wmma_f16(a, b, c);
    int row0 = rt * 16 + half * 8;
    int col  = ct * 16 + nl;
    size_t tok = (size_t)seq * L_ + t0 + col;
#pragma unroll
    for (int i = 0; i < 8; ++i) {
      float u2 = c[i] + (float)ucs[col][row0 + i];
      float gv = (float)gate[tok * HID_ + row0 + i];
      vs[col][row0 + i] = (_Float16)(u2 * gv);
    }
  }
  __syncthreads();

  // stage 2: grouped mix (4 row-tiles/group, K=64 -> 2 WMMA) -> ucs (reuse)
  for (int t = wave; t < 16 * 3; t += 8) {
    int rt = t / 3, ct = t % 3;
    int g = rt >> 2;
    v8f c = {};
#pragma unroll
    for (int ks = 0; ks < 2; ++ks) {
      v16h a = frag_a_f16g(Wmix, 64, rt * 16, ks * 32);
      v16h b = frag_b_ldsT(&vs[0][0], LD_HID, g * 64 + ks * 32, ct * 16);
      c = wmma_f16(a, b, c);
    }
    int row0 = rt * 16 + half * 8;
    int col  = ct * 16 + nl;
    float bias[8]; load_bias8(bmix + row0, bias);
#pragma unroll
    for (int i = 0; i < 8; ++i)
      ucs[col][row0 + i] = (_Float16)(c[i] + bias[i]);
  }
  __syncthreads();

  // stage 3: out_proj 128x256 (K=256 -> 8 WMMA)
  for (int t = wave; t < 8 * 3; t += 8) {
    int rt = t / 3, ct = t % 3;
    v8f c = {};
#pragma unroll
    for (int ks = 0; ks < HID_ / 32; ++ks) {
      v16h a = frag_a_f16g(Wop, HID_, rt * 16, ks * 32);
      v16h b = frag_b_ldsT(&ucs[0][0], LD_HID, ks * 32, ct * 16);
      c = wmma_f16(a, b, c);
    }
    int row0 = rt * 16 + half * 8;
    int col  = ct * 16 + nl;
    size_t tok = (size_t)seq * L_ + t0 + col;
    float bias[8]; load_bias8(bop + row0, bias);
#pragma unroll
    for (int i = 0; i < 8; ++i)
      yAll[tok * DIM_ + row0 + i] = c[i] + bias[i];
  }
}

// ---------------------------------------------------------------------------
// Kernel D: gather 4 direction outputs per pixel (permutations folded into
// addressing), fuse_w (128x512, K=512 -> 16 WMMA) + fuse_b.
// ---------------------------------------------------------------------------
__global__ void kD_fuse(const float* __restrict__ yAll,    // [NSEQ][L][DIM]
                        const _Float16* __restrict__ Wf,   // (128,512) f16
                        const float* __restrict__ bf,
                        float* __restrict__ out) {         // [B][DIM][L]
  const int tile = blockIdx.x, bi = blockIdx.y;
  const int t0 = tile * TN;
  __shared__ __align__(16) _Float16 ys[TN][LD_CAT];

  if (threadIdx.x == 0) { __builtin_prefetch(Wf, 0, 2); }

  for (int idx = threadIdx.x; idx < TN * 4 * DIM_; idx += blockDim.x) {
    int col = idx / (4 * DIM_), r = idx % (4 * DIM_);
    int dir = r >> 7, ch = r & (DIM_ - 1);
    int p = t0 + col;                       // row-major pixel index
    int tv = (p % W_) * H_ + (p / W_);      // v-ordering position of pixel p
    int tok = (dir == 0) ? p
            : (dir == 1) ? (L_ - 1 - p)
            : (dir == 2) ? tv
                         : (L_ - 1 - tv);
    int seq = dir * B_ + bi;
    ys[col][r] = (_Float16)yAll[((size_t)seq * L_ + tok) * DIM_ + ch];
  }
  __syncthreads();

  const int wave = threadIdx.x >> 5;
  const int lane = lane_id(), nl = lane & 15, half = lane >> 4;
  for (int t = wave; t < 8 * 3; t += 8) {
    int rt = t / 3, ct = t % 3;
    v8f c = {};
#pragma unroll
    for (int ks = 0; ks < (4 * DIM_) / 32; ++ks) {
      v16h a = frag_a_f16g(Wf, 4 * DIM_, rt * 16, ks * 32);
      v16h b = frag_b_ldsT(&ys[0][0], LD_CAT, ks * 32, ct * 16);
      c = wmma_f16(a, b, c);
    }
    int row0 = rt * 16 + half * 8;
    int col  = ct * 16 + nl;
    float bias[8]; load_bias8(bf + row0, bias);
#pragma unroll
    for (int i = 0; i < 8; ++i)
      out[((size_t)bi * DIM_ + row0 + i) * L_ + t0 + col] = c[i] + bias[i];
  }
}

// ---------------------------------------------------------------------------
// kernel_launch
// ---------------------------------------------------------------------------
extern "C" void kernel_launch(void* const* d_in, const int* in_sizes, int n_in,
                              void* d_out, int out_size, void* d_ws, size_t ws_size,
                              hipStream_t stream) {
  (void)in_sizes; (void)n_in; (void)out_size; (void)ws_size;

  const float* x          = (const float*)d_in[0];
  const float* in_proj_w  = (const float*)d_in[1];
  const float* in_proj_b  = (const float*)d_in[2];
  const float* dw_w       = (const float*)d_in[3];
  const float* ssm_in_w   = (const float*)d_in[4];
  const float* ssm_delta_w= (const float*)d_in[5];
  const float* ssm_delta_b= (const float*)d_in[6];
  const float* ssm_b_w    = (const float*)d_in[7];
  const float* ssm_b_b    = (const float*)d_in[8];
  const float* ssm_c_w    = (const float*)d_in[9];
  const float* ssm_c_b    = (const float*)d_in[10];
  const float* ssm_d      = (const float*)d_in[11];
  const float* ssm_out_w  = (const float*)d_in[12];
  const float* mix_w      = (const float*)d_in[13];
  const float* mix_b      = (const float*)d_in[14];
  const float* out_proj_w = (const float*)d_in[15];
  const float* out_proj_b = (const float*)d_in[16];
  const float* fuse_w     = (const float*)d_in[17];
  const float* fuse_b     = (const float*)d_in[18];
  float* outp = (float*)d_out;

  // Workspace carve-out (~397 MB total)
  const size_t nUH = (size_t)NSEQ * HID_ * L_;   // 37,748,736
  const size_t nUM = (size_t)NSEQ * MH_  * L_;   //  4,718,592
  const size_t nUD = (size_t)NSEQ * DIM_ * L_;   // 18,874,368
  char* p = (char*)d_ws;
  _Float16* upre = (_Float16*)p; p += nUH * 2;
  _Float16* ucB  = (_Float16*)p; p += nUH * 2;
  _Float16* gatB = (_Float16*)p; p += nUH * 2;
  float* sS  = (float*)p; p += nUM * 4;
  float* aS  = (float*)p; p += nUM * 4;
  float* bbS = (float*)p; p += nUM * 4;
  float* ccS = (float*)p; p += nUM * 4;
  float* gS  = (float*)p; p += nUM * 4;
  float* yAll= (float*)p; p += nUD * 4;
  // f16 weight copies (all segment sizes are multiples of 8 -> 16B aligned)
  _Float16* WpH   = (_Float16*)p; p += (size_t)512 * 32  * 2;
  _Float16* WiH   = (_Float16*)p; p += (size_t)32  * 256 * 2;
  _Float16* WdH   = (_Float16*)p; p += (size_t)32  * 256 * 2;
  _Float16* WbH   = (_Float16*)p; p += (size_t)32  * 256 * 2;
  _Float16* WcH   = (_Float16*)p; p += (size_t)32  * 256 * 2;
  _Float16* WoutH = (_Float16*)p; p += (size_t)256 * 32  * 2;
  _Float16* WmixH = (_Float16*)p; p += (size_t)256 * 64  * 2;
  _Float16* WopH  = (_Float16*)p; p += (size_t)128 * 256 * 2;
  _Float16* WfH   = (_Float16*)p; p += (size_t)128 * 512 * 2;

  const dim3 blk(256);
  const dim3 gridA(L_ / TN, NSEQ);   // 192 x 16
  const dim3 gridD(L_ / TN, B_);     // 192 x 4

  auto cvt = [&](const float* s, _Float16* d, int n) {
    kCvt<<<dim3((n + 255) / 256), blk, 0, stream>>>(s, d, n);
  };
  cvt(in_proj_w,  WpH,   512 * 32);
  cvt(ssm_in_w,   WiH,   32 * 256);
  cvt(ssm_delta_w,WdH,   32 * 256);
  cvt(ssm_b_w,    WbH,   32 * 256);
  cvt(ssm_c_w,    WcH,   32 * 256);
  cvt(ssm_out_w,  WoutH, 256 * 32);
  cvt(mix_w,      WmixH, 256 * 64);
  cvt(out_proj_w, WopH,  128 * 256);
  cvt(fuse_w,     WfH,   128 * 512);

  kA1_inproj<<<gridA, blk, 0, stream>>>(x, WpH, in_proj_b, upre, gatB);

  kA2_conv_ssm<<<gridA, blk, 0, stream>>>(upre, dw_w,
                                          WiH,
                                          WdH, ssm_delta_b,
                                          WbH, ssm_b_b,
                                          WcH, ssm_c_b,
                                          ucB, sS, aS, bbS, ccS);

  kB_scan<<<dim3((NSEQ * MH_ * 32) / 256), blk, 0, stream>>>(
      sS, aS, bbS, ccS, ssm_d, gS);

  kC_backend<<<gridA, blk, 0, stream>>>(ucB, gatB, gS,
                                        WoutH, WmixH, mix_b,
                                        WopH, out_proj_b, yAll);

  kD_fuse<<<gridD, blk, 0, stream>>>(yAll, WfH, fuse_b, outp);
}
